// Head_58978490909157
// MI455X (gfx1250) — matrix operand order repaired and checked by reading, not compiled, not run on hardware
//
#include <hip/hip_runtime.h>

typedef __attribute__((ext_vector_type(16))) _Float16 v16h;
typedef __attribute__((ext_vector_type(8)))  _Float16 v8h;
typedef __attribute__((ext_vector_type(4)))  _Float16 v4h;
typedef __attribute__((ext_vector_type(8)))  float    v8f;

union V16 { v16h v; v8h h[2]; };

__device__ __forceinline__ float sigmoidf(float x) {
    return 1.0f / (1.0f + __expf(-x));
}

// ---------------------------------------------------------------------------
// Pre-kernel: convert fp32 weights -> f16 and swizzle into the exact
// v_wmma_f32_16x16x32_f16 A-fragment layout (ISA 7.12.2, 16-bit A 16x32):
//   lane L holds row M = L&15
//   half index h (0..15): VGPR j = h/2, pos = h&1
//   K(L,h) = 32*S + ((j&4)?16:0) + 2*(j&3) + pos + 8*(L>>4)
// Packed layout: wp[(((t*(C/32)+S)*32 + L)*16 + h)]
//   -> per wave/K-step the fragment is 32 lanes x 32 contiguous bytes.
// ---------------------------------------------------------------------------
template<int C>
__global__ __launch_bounds__(256) void pack_w_kernel(const float* __restrict__ w,
                                                     _Float16* __restrict__ wp) {
    constexpr int SB = (C == 256) ? 3 : (C == 512) ? 4 : 5;   // log2(C/32)
    int idx = blockIdx.x * 256 + threadIdx.x;                 // < 256*C
    int h = idx & 15;
    int l = (idx >> 4) & 31;
    int s = (idx >> 9) & ((C / 32) - 1);
    int t = idx >> (9 + SB);
    int j = h >> 1, pos = h & 1;
    int k = s * 32 + ((j & 4) ? 16 : 0) + 2 * (j & 3) + pos + 8 * (l >> 4);
    int oc = t * 16 + (l & 15);
    float v = (oc < 255) ? w[(size_t)oc * C + k] : 0.0f;      // pad row 255
    wp[idx] = (_Float16)v;
}

// ---------------------------------------------------------------------------
// Main head kernel: one block = NT=4 pixel tiles (64 px, possibly spanning
// batch images), 16 waves; wave w owns oc-tile w and NT accumulators, so one
// A fragment feeds 4 WMMAs per K-step (4x less L2 weight traffic).
// K is chunked at KC=256 through LDS to bound shared memory at ~34 KB.
// ---------------------------------------------------------------------------
template<int C, int HW, int LVL_OFF, int LVL>
__global__ __launch_bounds__(512) void head_kernel(const float* __restrict__ f,
                                                   const _Float16* __restrict__ wpack,
                                                   const float* __restrict__ bias,
                                                   const float* __restrict__ anchors,
                                                   float* __restrict__ out) {
    constexpr int NT      = 4;                 // pixel tiles per block
    constexpr int KC      = 256;               // K chunk through LDS (C is always >= 256)
    constexpr int NCHUNK  = C / KC;
    constexpr int SPC     = KC / 32;           // K-steps per chunk
    constexpr int KSTEPS  = C / 32;
    constexpr int CP      = KC + 8;            // pad: 16B-aligned rows, no bank conflicts
    constexpr int TPB     = HW / 16;           // 16-px tiles per batch image
    constexpr int QPC     = KC / 4;            // channel-quads per chunk

    __shared__ __align__(16) _Float16 sf[NT * 16 * CP];   // 33792 B

    const int tile0 = blockIdx.x * NT;
    const int wave  = threadIdx.x >> 5;        // oc tile 0..15
    const int lane  = threadIdx.x & 31;

    int bIdx[NT], pBase[NT];
#pragma unroll
    for (int j = 0; j < NT; ++j) {
        int t    = tile0 + j;
        bIdx[j]  = t / TPB;
        pBase[j] = (t % TPB) * 16;
    }

    v8f acc[NT];
    v8f zero = {};
#pragma unroll
    for (int j = 0; j < NT; ++j) acc[j] = zero;

    const _Float16* wp0 = wpack + ((size_t)(wave * KSTEPS) * 32 + lane) * 16;

    for (int kc = 0; kc < NCHUNK; ++kc) {
        if (kc) __syncthreads();               // WAR: previous chunk fully consumed

        // Stage NT*16 pixels x KC channels into LDS as f16, [tilepx][KC+8].
        // Each thread: 4 consecutive channels of one pixel -> one ds_store_b64.
        // 16 consecutive lanes read 16 consecutive pixels of one channel row.
        for (int idx = threadIdx.x; idx < NT * 16 * QPC; idx += 512) {
            int px = idx & 15;
            int q  = (idx >> 4) & (QPC - 1);   // channel quad within chunk
            int j  = (idx >> 4) / QPC;         // tile
            int c  = q * 4;
            const float* g =
                f + (size_t)bIdx[j] * C * HW + (size_t)(kc * KC + c) * HW + pBase[j] + px;
            float v0 = __builtin_nontemporal_load(g);
            float v1 = __builtin_nontemporal_load(g + (size_t)HW);
            float v2 = __builtin_nontemporal_load(g + (size_t)2 * HW);
            float v3 = __builtin_nontemporal_load(g + (size_t)3 * HW);
            v4h hq = { (_Float16)v0, (_Float16)v1, (_Float16)v2, (_Float16)v3 };
            *(v4h*)(sf + (j * 16 + px) * CP + c) = hq;
        }
        __syncthreads();

        const _Float16* wp = wp0 + (size_t)(kc * SPC) * 512;
        for (int s = 0; s < SPC; ++s) {
            V16 a;
            a.h[0] = *(const v8h*)(wp + s * 512);
            a.h[1] = *(const v8h*)(wp + s * 512 + 8);

            // Preload all NT B fragments first so the ds-load latency of
            // fragment j overlaps the WMMAs of fragments < j.
            V16 bf[NT];
#pragma unroll
            for (int j = 0; j < NT; ++j) {
                // B (ISA 16-bit B 32x16): lane = column N = lane&15;
                // lanes 16-31 hold K=8..15 and K=24..31 of the 32-wide K step.
                const _Float16* bp =
                    sf + (j * 16 + (lane & 15)) * CP + (lane >> 4) * 8 + s * 32;
                bf[j].h[0] = *(const v8h*)(bp);
                bf[j].h[1] = *(const v8h*)(bp + 16);
            }
#pragma unroll
            for (int j = 0; j < NT; ++j) {
                acc[j] = __builtin_amdgcn_wmma_f32_16x16x32_f16(
                    /*neg_a=*/false, a.v, /*neg_b=*/false, bf[j].v,
                    /*c_mod=*/(short)0, acc[j], /*reuse_a=*/false, /*reuse_b=*/false);
            }
        }
    }

    // Epilogue: D layout: lane&15 = N (pixel), VGPR r -> M = r + 8*(lane>>4).
    const int ocbase = wave * 16 + 8 * (lane >> 4);
#pragma unroll
    for (int j = 0; j < NT; ++j) {
        const int    p    = pBase[j] + (lane & 15);
        const size_t rowb = (size_t)bIdx[j] * 25200 + LVL_OFF + p;
#pragma unroll
        for (int r = 0; r < 8; ++r) {
            int oc = ocbase + r;
            if (oc < 255) {
                float v = acc[j][r] + bias[oc];
                int ai = oc / 85;              // anchor index
                int e  = oc - ai * 85;         // element within anchor
                if (e < 2) {
                    v = 2.0f * sigmoidf(v) - 0.5f;                      // xy
                } else if (e < 4) {
                    float t2 = 2.0f * sigmoidf(v);
                    v = t2 * t2 * anchors[LVL * 6 + ai * 2 + (e - 2)];  // wh
                }
                __builtin_nontemporal_store(v, out + (rowb + (size_t)ai * HW) * 85 + e);
            }
        }
    }
}

extern "C" void kernel_launch(void* const* d_in, const int* in_sizes, int n_in,
                              void* d_out, int out_size, void* d_ws, size_t ws_size,
                              hipStream_t stream) {
    (void)in_sizes; (void)n_in; (void)out_size; (void)ws_size;

    const float* f0 = (const float*)d_in[0];   // (16, 256, 80, 80)
    const float* f1 = (const float*)d_in[1];   // (16, 512, 40, 40)
    const float* f2 = (const float*)d_in[2];   // (16,1024, 20, 20)
    const float* w0 = (const float*)d_in[3];   // (255, 256)
    const float* b0 = (const float*)d_in[4];
    const float* w1 = (const float*)d_in[5];   // (255, 512)
    const float* b1 = (const float*)d_in[6];
    const float* w2 = (const float*)d_in[7];   // (255,1024)
    const float* b2 = (const float*)d_in[8];
    const float* an = (const float*)d_in[9];   // (3,3,2)
    float* out = (float*)d_out;

    // Packed f16 weights in workspace: 256*C halves per level (~0.9 MB total).
    _Float16* wp = (_Float16*)d_ws;
    const size_t off0 = 0;
    const size_t off1 = off0 + 256 * 256;
    const size_t off2 = off1 + 256 * 512;

    pack_w_kernel< 256><<<(256 *  256) / 256, 256, 0, stream>>>(w0, wp + off0);
    pack_w_kernel< 512><<<(256 *  512) / 256, 256, 0, stream>>>(w1, wp + off1);
    pack_w_kernel<1024><<<(256 * 1024) / 256, 256, 0, stream>>>(w2, wp + off2);

    // Blocks = (B * HW / 16) / NT tiles; NT = 4.
    // L0: 16*400/4 = 1600 ; L1: 16*100/4 = 400 ; L2: 16*25/4 = 100.
    head_kernel< 256, 6400,     0, 0><<<1600, 512, 0, stream>>>(f0, wp + off0, b0, an, out);
    head_kernel< 512, 1600, 19200, 1><<< 400, 512, 0, stream>>>(f1, wp + off1, b1, an, out);
    head_kernel<1024,  400, 24000, 2><<< 100, 512, 0, stream>>>(f2, wp + off2, b2, an, out);
}
